// SelfAttV1_1348619731355
// MI455X (gfx1250) — compile-verified, matrix operand order checked by reading, not run on hardware
//
#include <hip/hip_runtime.h>
#include <math.h>

// ---------------------------------------------------------------------------
// Self-attention forward, B=4, S=2048, H=1024 (f32 in/out) on gfx1250.
// All GEMMs run on V_WMMA_F32_16X16X32_BF16 with an async-copy double-buffered
// LDS pipeline (GLOBAL_LOAD_ASYNC_TO_LDS_B128 + s_wait_asynccnt).
//
// Pipeline (all on `stream`):
//   0) cvt_kernel    : X, Wq, Wk, Wv  f32 -> bf16 copies in ws
//   1) qkv_kernel    : Q = Xb Wqb^T + bq, K = Xb Wkb^T + bk  (bf16, [B*S,H])
//                      V^T = (Xb Wvb^T + bv)^T               (bf16, [B,H,S])
//   2) scores_kernel : Sc = (Q K^T) / sqrt(H)                (f32,  [B,S,S])
//   3) softmax_kernel: At = softmax_rows(Sc)                 (bf16, [B,S,S])
//   4) out_kernel    : out = At V = At (V^T)^T               (f32,  d_out)
//
// Workspace layout (166 MB total):
//   [0,16)    Xb bf16         [16,22)   Wq/Wk/Wv bf16 (2 MB each)
//   [22,38)   Q bf16          [38,54)   K bf16        [54,70)  V^T bf16
//   [70,134)  scores f32      [134,166) attn bf16
// ---------------------------------------------------------------------------

typedef __attribute__((ext_vector_type(16))) __bf16 v16bf;
typedef __attribute__((ext_vector_type(8)))  float  v8f;

#define TB   256   // threads per block = 8 waves (wave32)
#define BM   128   // block tile M
#define BN   128   // block tile N
#define BK   32    // block tile K (one WMMA K-step)
#define LDSS 40    // LDS row stride in bf16 elems (32 + 8 pad -> 80B rows)

static constexpr int Hdim = 1024;
static constexpr int Sdim = 2048;
static constexpr int Bdim = 4;
static constexpr int Mtot = Bdim * Sdim;   // 8192 tokens

// round-to-nearest-even f32 -> bf16
__device__ __forceinline__ __bf16 f2bf(float x) {
  union { float f; unsigned u; } c; c.f = x;
  unsigned r = (c.u + 0x7FFFu + ((c.u >> 16) & 1u)) >> 16;
  union { unsigned short s; __bf16 b; } o; o.s = (unsigned short)r;
  return o.b;
}

__device__ __forceinline__ v8f wmma_bf16(v16bf a, v16bf b, v8f c) {
  // (neg_a, A, neg_b, B, c_mod, C, reuse_a, reuse_b)
  return __builtin_amdgcn_wmma_f32_16x16x32_bf16(
      false, a, false, b, (short)0, c, false, false);
}

// ---- CDNA5 async global->LDS copy (ASYNCcnt-tracked, bypasses VGPRs) ------
// Flat/generic pointers into LDS carry the LDS byte offset in addr[31:0]
// (ISA 10.2), so truncating the generic pointer
// gives the VDST LDS-address operand.
__device__ __forceinline__ void async_cp_b128(void* lds_ptr, const void* gptr) {
  const unsigned loff = (unsigned)(unsigned long long)lds_ptr;
  asm volatile("global_load_async_to_lds_b128 %0, %1, off"
               :: "v"(loff), "v"(gptr) : "memory");
}

__device__ __forceinline__ void wait_async0() {
  asm volatile("s_wait_asynccnt 0" ::: "memory");
}

// Stage one 128x32 bf16 tile into LDS with async copies: each of 256 threads
// issues two b128 async loads (row = t/2, 16-elem chunk = t%2).
__device__ __forceinline__ void stage_async(const __bf16* g, int ld, __bf16* s) {
  const int t   = threadIdx.x;
  const int row = t >> 1;            // 0..127
  const int kc  = (t & 1) << 4;      // 0 or 16 (elements)
  const __bf16* gp = g + (size_t)row * ld + kc;
  __bf16* sp = s + row * LDSS + kc;
  async_cp_b128(sp,     gp);         // 16 B
  async_cp_b128(sp + 8, gp + 8);     // 16 B
}

// Load one 16x32 bf16 WMMA operand fragment from LDS (row-major, stride LDSS).
// ISA 16-bit A/B layout: lanes 0-15 hold rows 0-15 with K=0..7 & 16..23,
// lanes 16-31 hold same rows with K offset by 8. Two 16B chunks per lane.
__device__ __forceinline__ v16bf load_frag(const __bf16* base) {
  const int lane = threadIdx.x & 31;
  const __bf16* p = base + (lane & 15) * LDSS + ((lane >> 4) << 3);
  v16bf f;
  uint4* fv = reinterpret_cast<uint4*>(&f);
  fv[0] = *reinterpret_cast<const uint4*>(p);        // K chunk 0
  fv[1] = *reinterpret_cast<const uint4*>(p + 16);   // K chunk 1
  return f;
}

// NT GEMM core with double-buffered async LDS pipeline.
// C[128x128] += A[128xK] * B[128xK]^T, both operands bf16 row-major over the
// contraction dim. 8 waves in a 4x2 grid; each wave owns a 32x64 macro-tile.
__device__ __forceinline__ void gemm_core(const __bf16* A, int lda,
                                          const __bf16* B, int ldb,
                                          int K, v8f acc[2][4]) {
  __shared__ __bf16 As[2][BM * LDSS];
  __shared__ __bf16 Bs[2][BN * LDSS];
  const int wave = threadIdx.x >> 5;
  const int wm = wave >> 1;     // 0..3
  const int wn = wave & 1;      // 0..1

  // Prologue: fill buffer 0.
  stage_async(A, lda, As[0]);
  stage_async(B, ldb, Bs[0]);
  wait_async0();
  __syncthreads();

  int cur = 0;
  for (int k0 = 0; k0 < K; k0 += BK) {
    const int nxt = cur ^ 1;
    if (k0 + BK < K) {                        // prefetch next K-tile
      stage_async(A + (k0 + BK), lda, As[nxt]);
      stage_async(B + (k0 + BK), ldb, Bs[nxt]);
    }
    v16bf a0 = load_frag(As[cur] + (wm * 32 +  0) * LDSS);
    v16bf a1 = load_frag(As[cur] + (wm * 32 + 16) * LDSS);
    v16bf b0 = load_frag(Bs[cur] + (wn * 64 +  0) * LDSS);
    v16bf b1 = load_frag(Bs[cur] + (wn * 64 + 16) * LDSS);
    v16bf b2 = load_frag(Bs[cur] + (wn * 64 + 32) * LDSS);
    v16bf b3 = load_frag(Bs[cur] + (wn * 64 + 48) * LDSS);
    acc[0][0] = wmma_bf16(a0, b0, acc[0][0]);
    acc[0][1] = wmma_bf16(a0, b1, acc[0][1]);
    acc[0][2] = wmma_bf16(a0, b2, acc[0][2]);
    acc[0][3] = wmma_bf16(a0, b3, acc[0][3]);
    acc[1][0] = wmma_bf16(a1, b0, acc[1][0]);
    acc[1][1] = wmma_bf16(a1, b1, acc[1][1]);
    acc[1][2] = wmma_bf16(a1, b2, acc[1][2]);
    acc[1][3] = wmma_bf16(a1, b3, acc[1][3]);
    wait_async0();       // my async writes into buf[nxt] complete
    __syncthreads();     // all waves: nxt fully written, cur fully read
    cur = nxt;
  }
}

// Per-lane position of element r of accumulator (i,j) inside the 128x128 tile.
// f32 C/D layout: lane = N (0-15 / 16-31), VGPR r = M row r, lanes>=16 -> M+8.
struct TilePos { int m, n; };
__device__ __forceinline__ TilePos tile_pos(int i, int j) {
  const int lane = threadIdx.x & 31;
  const int wave = threadIdx.x >> 5;
  TilePos p;
  p.m = (wave >> 1) * 32 + i * 16 + ((lane >> 4) << 3);
  p.n = (wave &  1) * 64 + j * 16 + (lane & 15);
  return p;
}

// ---------------------------------------------------------------------------
// Kernel 0: f32 -> bf16 conversion (bandwidth-bound, ~3 us total).
// ---------------------------------------------------------------------------
__global__ __launch_bounds__(TB) void cvt_kernel(const float* __restrict__ src,
                                                 __bf16* __restrict__ dst,
                                                 int n4) {   // n/4 float4 groups
  const int i = blockIdx.x * TB + threadIdx.x;
  if (i < n4) {
    const float4 f = reinterpret_cast<const float4*>(src)[i];
    __bf16 o[4] = { f2bf(f.x), f2bf(f.y), f2bf(f.z), f2bf(f.w) };
    reinterpret_cast<uint2*>(dst)[i] = *reinterpret_cast<uint2*>(o);
  }
}

// ---------------------------------------------------------------------------
// Kernel 1: fused QKV projection (bf16 inputs). blockIdx.z selects Q / K / V.
// ---------------------------------------------------------------------------
__global__ __launch_bounds__(TB) void qkv_kernel(
    const __bf16* __restrict__ Xb,
    const __bf16* __restrict__ Wqb, const float* __restrict__ bq,
    const __bf16* __restrict__ Wkb, const float* __restrict__ bk,
    const __bf16* __restrict__ Wvb, const float* __restrict__ bv,
    __bf16* __restrict__ Q, __bf16* __restrict__ Kc, __bf16* __restrict__ Vt) {
  const int which = blockIdx.z;
  const __bf16* W   = (which == 0) ? Wqb : (which == 1) ? Wkb : Wvb;
  const float* bias = (which == 0) ? bq  : (which == 1) ? bk  : bv;
  const int m0 = blockIdx.x * BM;    // token block
  const int n0 = blockIdx.y * BN;    // feature block
  v8f acc[2][4] = {};
  gemm_core(Xb + (size_t)m0 * Hdim, Hdim,
            W  + (size_t)n0 * Hdim, Hdim, Hdim, acc);
  if (which < 2) {
    __bf16* O = (which == 0) ? Q : Kc;
#pragma unroll
    for (int i = 0; i < 2; ++i)
#pragma unroll
      for (int j = 0; j < 4; ++j) {
        const TilePos p = tile_pos(i, j);
        const int n = n0 + p.n;
        const float bb = bias[n];
#pragma unroll
        for (int r = 0; r < 8; ++r)
          O[(size_t)(m0 + p.m + r) * Hdim + n] = f2bf(acc[i][j][r] + bb);
      }
  } else {
#pragma unroll
    for (int i = 0; i < 2; ++i)
#pragma unroll
      for (int j = 0; j < 4; ++j) {
        const TilePos p = tile_pos(i, j);
        const int n = n0 + p.n;
        const float bb = bias[n];
#pragma unroll
        for (int r = 0; r < 8; ++r) {
          const int m   = m0 + p.m + r;
          const int bat = m >> 11;           // m / Sdim
          const int s   = m & (Sdim - 1);
          Vt[((size_t)bat * Hdim + n) * Sdim + s] = f2bf(acc[i][j][r] + bb);
        }
      }
  }
}

// ---------------------------------------------------------------------------
// Kernel 2: scores = Q K^T / sqrt(H), per batch (blockIdx.z).
// ---------------------------------------------------------------------------
__global__ __launch_bounds__(TB) void scores_kernel(
    const __bf16* __restrict__ Q, const __bf16* __restrict__ Kc,
    float* __restrict__ Sc) {
  const int bat = blockIdx.z;
  const int m0 = blockIdx.x * BM;
  const int n0 = blockIdx.y * BN;
  const __bf16* Qb = Q  + (size_t)bat * Sdim * Hdim;
  const __bf16* Kb = Kc + (size_t)bat * Sdim * Hdim;
  float* Sb = Sc + (size_t)bat * Sdim * Sdim;
  v8f acc[2][4] = {};
  gemm_core(Qb + (size_t)m0 * Hdim, Hdim,
            Kb + (size_t)n0 * Hdim, Hdim, Hdim, acc);
  const float scale = 0.03125f;   // 1/sqrt(1024)
#pragma unroll
  for (int i = 0; i < 2; ++i)
#pragma unroll
    for (int j = 0; j < 4; ++j) {
      const TilePos p = tile_pos(i, j);
#pragma unroll
      for (int r = 0; r < 8; ++r)
        Sb[(size_t)(m0 + p.m + r) * Sdim + (n0 + p.n)] = acc[i][j][r] * scale;
    }
}

// ---------------------------------------------------------------------------
// Kernel 3: row softmax over 2048 f32 -> bf16. One block per (b, q) row.
// ---------------------------------------------------------------------------
__global__ __launch_bounds__(TB) void softmax_kernel(
    const float* __restrict__ Sc, __bf16* __restrict__ At) {
  __shared__ float red[TB];
  const size_t row = blockIdx.x;            // 0 .. B*S-1
  const float* r = Sc + row * Sdim;
  __bf16* o = At + row * Sdim;
  const int t = threadIdx.x;
  float v[8];
  float mx = -3.402823466e38f;
#pragma unroll
  for (int i = 0; i < 8; ++i) { v[i] = r[t + i * TB]; mx = fmaxf(mx, v[i]); }
  red[t] = mx; __syncthreads();
  for (int s = TB / 2; s > 0; s >>= 1) {
    if (t < s) red[t] = fmaxf(red[t], red[t + s]);
    __syncthreads();
  }
  mx = red[0]; __syncthreads();
  float sum = 0.f;
#pragma unroll
  for (int i = 0; i < 8; ++i) { v[i] = __expf(v[i] - mx); sum += v[i]; }
  red[t] = sum; __syncthreads();
  for (int s = TB / 2; s > 0; s >>= 1) {
    if (t < s) red[t] += red[t + s];
    __syncthreads();
  }
  const float inv = 1.f / red[0];
#pragma unroll
  for (int i = 0; i < 8; ++i) o[t + i * TB] = f2bf(v[i] * inv);
}

// ---------------------------------------------------------------------------
// Kernel 4: out = attn @ V  (V stored transposed -> same NT GEMM pattern).
// ---------------------------------------------------------------------------
__global__ __launch_bounds__(TB) void out_kernel(
    const __bf16* __restrict__ At, const __bf16* __restrict__ Vt,
    float* __restrict__ O) {
  const int bat = blockIdx.z;
  const int m0 = blockIdx.x * BM;
  const int n0 = blockIdx.y * BN;
  const __bf16* Ab = At + (size_t)bat * Sdim * Sdim;
  const __bf16* Vb = Vt + (size_t)bat * Hdim * Sdim;
  float* Ob = O + (size_t)bat * Sdim * Hdim;
  v8f acc[2][4] = {};
  gemm_core(Ab + (size_t)m0 * Sdim, Sdim,
            Vb + (size_t)n0 * Sdim, Sdim, Sdim, acc);
#pragma unroll
  for (int i = 0; i < 2; ++i)
#pragma unroll
    for (int j = 0; j < 4; ++j) {
      const TilePos p = tile_pos(i, j);
#pragma unroll
      for (int r = 0; r < 8; ++r)
        Ob[(size_t)(m0 + p.m + r) * Hdim + (n0 + p.n)] = acc[i][j][r];
    }
}

// ---------------------------------------------------------------------------
extern "C" void kernel_launch(void* const* d_in, const int* in_sizes, int n_in,
                              void* d_out, int out_size, void* d_ws, size_t ws_size,
                              hipStream_t stream) {
  const float* X  = (const float*)d_in[0];
  const float* Wq = (const float*)d_in[1];
  const float* bq = (const float*)d_in[2];
  const float* Wk = (const float*)d_in[3];
  const float* bk = (const float*)d_in[4];
  const float* Wv = (const float*)d_in[5];
  const float* bv = (const float*)d_in[6];
  float* out = (float*)d_out;

  char* ws = (char*)d_ws;
  const size_t MB = 1ull << 20;
  __bf16* Xb  = (__bf16*)(ws);               // 16 MB
  __bf16* Wqb = (__bf16*)(ws +  16 * MB);    //  2 MB
  __bf16* Wkb = (__bf16*)(ws +  18 * MB);    //  2 MB
  __bf16* Wvb = (__bf16*)(ws +  20 * MB);    //  2 MB
  __bf16* Q   = (__bf16*)(ws +  22 * MB);    // 16 MB
  __bf16* Kc  = (__bf16*)(ws +  38 * MB);    // 16 MB
  __bf16* Vt  = (__bf16*)(ws +  54 * MB);    // 16 MB  (V transposed, [B,H,S])
  float*  Sc  = (float*) (ws +  70 * MB);    // 64 MB  (scores f32)
  __bf16* At  = (__bf16*)(ws + 134 * MB);    // 32 MB  (attn bf16)

  const dim3 blk(TB, 1, 1);
  const int nX = Mtot * Hdim;                // 8,388,608
  const int nW = Hdim * Hdim;                // 1,048,576
  cvt_kernel<<<dim3((nX / 4 + TB - 1) / TB), blk, 0, stream>>>(X,  Xb,  nX / 4);
  cvt_kernel<<<dim3((nW / 4 + TB - 1) / TB), blk, 0, stream>>>(Wq, Wqb, nW / 4);
  cvt_kernel<<<dim3((nW / 4 + TB - 1) / TB), blk, 0, stream>>>(Wk, Wkb, nW / 4);
  cvt_kernel<<<dim3((nW / 4 + TB - 1) / TB), blk, 0, stream>>>(Wv, Wvb, nW / 4);
  qkv_kernel<<<dim3(Mtot / BM, Hdim / BN, 3), blk, 0, stream>>>(
      Xb, Wqb, bq, Wkb, bk, Wvb, bv, Q, Kc, Vt);
  scores_kernel<<<dim3(Sdim / BM, Sdim / BN, Bdim), blk, 0, stream>>>(Q, Kc, Sc);
  softmax_kernel<<<dim3(Bdim * Sdim, 1, 1), blk, 0, stream>>>(Sc, At);
  out_kernel<<<dim3(Sdim / BM, Hdim / BN, Bdim), blk, 0, stream>>>(At, Vt, out);
}
